// GroupedExperts_9912784519934
// MI455X (gfx1250) — matrix operand order repaired and checked by reading, not compiled, not run on hardware
//
#include <hip/hip_runtime.h>
#include <hip/hip_bf16.h>

// Problem constants (from reference)
#define E_   8
#define T_   4096
#define DIN_ 2048
#define DH_  1408

// Tiling
#define BM 128      // block tile M
#define BN 128      // block tile N
#define KB 32       // K step per WMMA (bf16 16x16x32)
#define KPAD 40     // padded LDS row (elements) -> 80B rows, 16B aligned
#define NK1 (DIN_ / KB)   // 64
#define NK2 (DH_ / KB)    // 44

typedef __attribute__((ext_vector_type(16))) __bf16 bf16x16;
typedef __attribute__((ext_vector_type(8)))  float  f32x8;

union Frag16 {
    bf16x16 v;
    uint4   q[2];
};

// Hardware f32->bf16 convert (let clang pick v_cvt_*_bf16_*)
__device__ __forceinline__ unsigned short to_bf16(float f) {
    __bf16 h = (__bf16)f;
    return __builtin_bit_cast(unsigned short, h);
}
__device__ __forceinline__ unsigned int pack_bf16(float lo, float hi) {
    return (unsigned int)to_bf16(lo) | ((unsigned int)to_bf16(hi) << 16);
}

// ---- CDNA5 async global->LDS copy (ASYNCcnt-tracked) ----------------------
__device__ __forceinline__ unsigned lds_offset(const void* p) {
    // generic -> LDS addrspace -> 32-bit LDS byte offset
    return (unsigned)(unsigned long long)
        (const __attribute__((address_space(3))) void*)p;
}
__device__ __forceinline__ void async_copy16(unsigned lds, const void* gaddr) {
    asm volatile("global_load_async_to_lds_b128 %0, %1, off"
                 :: "v"(lds), "v"(gaddr) : "memory");
}
#define S_WAIT_ASYNCCNT(NSTR) asm volatile("s_wait_asynccnt " NSTR ::: "memory")

// ---------------------------------------------------------------------------
// Pre-pass 1: x f32 -> bf16 straight copy.  8 elements / thread, no tail.
// ---------------------------------------------------------------------------
__global__ __launch_bounds__(256)
void cvt_x_kernel(const float* __restrict__ src, unsigned short* __restrict__ dst) {
    long long i = ((long long)blockIdx.x * 256 + threadIdx.x) * 8;
    float4 a = *(const float4*)(src + i);
    float4 b = *(const float4*)(src + i + 4);
    uint4 o;
    o.x = pack_bf16(a.x, a.y);
    o.y = pack_bf16(a.z, a.w);
    o.z = pack_bf16(b.x, b.y);
    o.w = pack_bf16(b.z, b.w);
    *(uint4*)(dst + i) = o;
}

// ---------------------------------------------------------------------------
// Pre-pass 2: transpose + convert.  src [E][R][C] f32 -> dst [E][C][R] bf16.
// grid: (C/32, R/32, E), block 256 (32x8)
// ---------------------------------------------------------------------------
__global__ __launch_bounds__(256)
void transp_cvt_kernel(const float* __restrict__ src,
                       unsigned short* __restrict__ dst, int R, int C) {
    __shared__ float tile[32][33];
    const int e  = blockIdx.z;
    const int c0 = blockIdx.x * 32;
    const int r0 = blockIdx.y * 32;
    const int tx = threadIdx.x & 31;
    const int ty = threadIdx.x >> 5;
    const float* s = src + (size_t)e * R * C;
    unsigned short* d = dst + (size_t)e * R * C;
#pragma unroll
    for (int i = 0; i < 4; ++i)
        tile[ty + 8 * i][tx] = s[(size_t)(r0 + ty + 8 * i) * C + c0 + tx];
    __syncthreads();
#pragma unroll
    for (int i = 0; i < 4; ++i)
        d[(size_t)(c0 + ty + 8 * i) * R + r0 + tx] = to_bf16(tile[tx][ty + 8 * i]);
}

// ---------------------------------------------------------------------------
// Kernel 1: h = silu(x@gate) * (x@down).  All operands pre-converted bf16.
//   xb [E][T][DIN], gT/dT [E][DH][DIN] (transposed), h [E][T][DH] bf16.
// grid: (DH/BN=11, T/BM=32, E), block 256
// ---------------------------------------------------------------------------
__global__ __launch_bounds__(256)
void moe_gatedown_kernel(const unsigned short* __restrict__ xb,
                         const unsigned short* __restrict__ gT,
                         const unsigned short* __restrict__ dT,
                         unsigned short* __restrict__ hws) {
    __shared__ alignas(16) unsigned short sA[2][BM][KPAD];
    __shared__ alignas(16) unsigned short sG[2][BN][KPAD];
    __shared__ alignas(16) unsigned short sD[2][BN][KPAD];

    const int e   = blockIdx.z;
    const int tm0 = blockIdx.y * BM;
    const int tn0 = blockIdx.x * BN;
    const int tid = threadIdx.x;
    const int lane = tid & 31;
    const int w    = tid >> 5;
    const int wm   = w >> 1;             // 0..3
    const int wn   = w & 1;              // 0..1
    const int lhi  = lane >> 4;
    const int llo  = lane & 15;

    const unsigned short* xe = xb + ((size_t)e * T_ + tm0) * DIN_;
    const unsigned short* ge = gT + ((size_t)e * DH_ + tn0) * DIN_;
    const unsigned short* de = dT + ((size_t)e * DH_ + tn0) * DIN_;

    // two 16B chunks per thread per 128x32 tile
    const int elem0 = tid * 8;
    const int elem1 = (tid + 256) * 8;
    const int r0c = elem0 >> 5, kk0 = elem0 & 31;
    const int r1c = elem1 >> 5, kk1 = elem1 & 31;

    auto stage = [&](int b, int k0) {
        async_copy16(lds_offset(&sA[b][r0c][kk0]), xe + (size_t)r0c * DIN_ + k0 + kk0);
        async_copy16(lds_offset(&sA[b][r1c][kk1]), xe + (size_t)r1c * DIN_ + k0 + kk1);
        async_copy16(lds_offset(&sG[b][r0c][kk0]), ge + (size_t)r0c * DIN_ + k0 + kk0);
        async_copy16(lds_offset(&sG[b][r1c][kk1]), ge + (size_t)r1c * DIN_ + k0 + kk1);
        async_copy16(lds_offset(&sD[b][r0c][kk0]), de + (size_t)r0c * DIN_ + k0 + kk0);
        async_copy16(lds_offset(&sD[b][r1c][kk1]), de + (size_t)r1c * DIN_ + k0 + kk1);
    };

    f32x8 accG[2][4], accD[2][4];
#pragma unroll
    for (int s = 0; s < 2; ++s)
#pragma unroll
        for (int u = 0; u < 4; ++u) { accG[s][u] = (f32x8)0.f; accD[s][u] = (f32x8)0.f; }

    stage(0, 0);
    for (int ks = 0; ks < NK1; ++ks) {
        const int b = ks & 1;
        if (ks + 1 < NK1) {
            stage((ks + 1) & 1, (ks + 1) * KB);
            S_WAIT_ASYNCCNT("6");   // in-order completion: previous batch done
        } else {
            S_WAIT_ASYNCCNT("0");
        }
        __syncthreads();

        Frag16 a[2];
#pragma unroll
        for (int s = 0; s < 2; ++s) {
            const unsigned short* base = &sA[b][wm * 32 + s * 16 + llo][0];
            a[s].q[0] = *(const uint4*)(base + lhi * 16);
            a[s].q[1] = *(const uint4*)(base + lhi * 16 + 8);
        }
#pragma unroll
        for (int u = 0; u < 4; ++u) {
            const int cn = wn * 64 + u * 16 + llo;
            Frag16 bg, bd;
            bg.q[0] = *(const uint4*)&sG[b][cn][lhi * 16];
            bg.q[1] = *(const uint4*)&sG[b][cn][lhi * 16 + 8];
            bd.q[0] = *(const uint4*)&sD[b][cn][lhi * 16];
            bd.q[1] = *(const uint4*)&sD[b][cn][lhi * 16 + 8];
#pragma unroll
            for (int s = 0; s < 2; ++s) {
                accG[s][u] = __builtin_amdgcn_wmma_f32_16x16x32_bf16(
                    false, a[s].v, false, bg.v, (short)0, accG[s][u], false, false);
                accD[s][u] = __builtin_amdgcn_wmma_f32_16x16x32_bf16(
                    false, a[s].v, false, bd.v, (short)0, accD[s][u], false, false);
            }
        }
        __syncthreads();
    }

    // epilogue: silu(g)*d -> bf16
    unsigned short* hb = hws + ((size_t)e * T_ + tm0) * DH_ + tn0;
#pragma unroll
    for (int s = 0; s < 2; ++s)
#pragma unroll
        for (int u = 0; u < 4; ++u) {
            const int n = wn * 64 + u * 16 + llo;
#pragma unroll
            for (int r = 0; r < 8; ++r) {
                const int m = wm * 32 + s * 16 + lhi * 8 + r;
                float g = accG[s][u][r];
                float d = accD[s][u][r];
                float t = 1.f + __expf(-g);
#if __has_builtin(__builtin_amdgcn_rcpf)
                float hv = g * __builtin_amdgcn_rcpf(t) * d;
#else
                float hv = g / t * d;
#endif
                hb[(size_t)m * DH_ + n] = to_bf16(hv);
            }
        }
}

// ---------------------------------------------------------------------------
// Kernel 2: out = h @ up.  h [E][T][DH] bf16, uT [E][DIN][DH] bf16 (transposed)
// grid: (DIN/BN=16, T/BM=32, E), block 256
// ---------------------------------------------------------------------------
__global__ __launch_bounds__(256)
void moe_up_kernel(const unsigned short* __restrict__ hws,
                   const unsigned short* __restrict__ uT,
                   float* __restrict__ out) {
    __shared__ alignas(16) unsigned short sA[2][BM][KPAD];
    __shared__ alignas(16) unsigned short sU[2][BN][KPAD];

    const int e   = blockIdx.z;
    const int tm0 = blockIdx.y * BM;
    const int tn0 = blockIdx.x * BN;
    const int tid = threadIdx.x;
    const int lane = tid & 31;
    const int w    = tid >> 5;
    const int wm   = w >> 1;
    const int wn   = w & 1;
    const int lhi  = lane >> 4;
    const int llo  = lane & 15;

    const unsigned short* he = hws + ((size_t)e * T_ + tm0) * DH_;
    const unsigned short* ue = uT + ((size_t)e * DIN_ + tn0) * DH_;

    const int elem0 = tid * 8;
    const int elem1 = (tid + 256) * 8;
    const int r0c = elem0 >> 5, kk0 = elem0 & 31;
    const int r1c = elem1 >> 5, kk1 = elem1 & 31;

    auto stage = [&](int b, int k0) {
        async_copy16(lds_offset(&sA[b][r0c][kk0]), he + (size_t)r0c * DH_ + k0 + kk0);
        async_copy16(lds_offset(&sA[b][r1c][kk1]), he + (size_t)r1c * DH_ + k0 + kk1);
        async_copy16(lds_offset(&sU[b][r0c][kk0]), ue + (size_t)r0c * DH_ + k0 + kk0);
        async_copy16(lds_offset(&sU[b][r1c][kk1]), ue + (size_t)r1c * DH_ + k0 + kk1);
    };

    f32x8 acc[2][4];
#pragma unroll
    for (int s = 0; s < 2; ++s)
#pragma unroll
        for (int u = 0; u < 4; ++u) acc[s][u] = (f32x8)0.f;

    stage(0, 0);
    for (int ks = 0; ks < NK2; ++ks) {
        const int b = ks & 1;
        if (ks + 1 < NK2) {
            stage((ks + 1) & 1, (ks + 1) * KB);
            S_WAIT_ASYNCCNT("4");
        } else {
            S_WAIT_ASYNCCNT("0");
        }
        __syncthreads();

        Frag16 a[2];
#pragma unroll
        for (int s = 0; s < 2; ++s) {
            const unsigned short* base = &sA[b][wm * 32 + s * 16 + llo][0];
            a[s].q[0] = *(const uint4*)(base + lhi * 16);
            a[s].q[1] = *(const uint4*)(base + lhi * 16 + 8);
        }
#pragma unroll
        for (int u = 0; u < 4; ++u) {
            const int cn = wn * 64 + u * 16 + llo;
            Frag16 bu;
            bu.q[0] = *(const uint4*)&sU[b][cn][lhi * 16];
            bu.q[1] = *(const uint4*)&sU[b][cn][lhi * 16 + 8];
#pragma unroll
            for (int s = 0; s < 2; ++s) {
                acc[s][u] = __builtin_amdgcn_wmma_f32_16x16x32_bf16(
                    false, a[s].v, false, bu.v, (short)0, acc[s][u], false, false);
            }
        }
        __syncthreads();
    }

    float* ob = out + ((size_t)e * T_ + tm0) * DIN_ + tn0;
#pragma unroll
    for (int s = 0; s < 2; ++s)
#pragma unroll
        for (int u = 0; u < 4; ++u) {
            const int n = wn * 64 + u * 16 + llo;
#pragma unroll
            for (int r = 0; r < 8; ++r) {
                const int m = wm * 32 + s * 16 + lhi * 8 + r;
                ob[(size_t)m * DIN_ + n] = acc[s][u][r];
            }
        }
}

// ---------------------------------------------------------------------------
extern "C" void kernel_launch(void* const* d_in, const int* in_sizes, int n_in,
                              void* d_out, int out_size, void* d_ws, size_t ws_size,
                              hipStream_t stream) {
    (void)in_sizes; (void)n_in; (void)out_size; (void)ws_size;
    const float* x    = (const float*)d_in[0];   // [E, T, DIN]
    const float* gate = (const float*)d_in[1];   // [E, DIN, DH]
    const float* down = (const float*)d_in[2];   // [E, DIN, DH]
    const float* up   = (const float*)d_in[3];   // [E, DH, DIN]
    float* out = (float*)d_out;                  // [E, T, DIN]

    // Workspace layout (bf16 elements): ~348 MB total
    const size_t XN = (size_t)E_ * T_ * DIN_;    // 67.1M
    const size_t WN = (size_t)E_ * DIN_ * DH_;   // 23.1M (each weight)
    unsigned short* xb = (unsigned short*)d_ws;
    unsigned short* gT = xb + XN;
    unsigned short* dT = gT + WN;
    unsigned short* uT = dT + WN;
    unsigned short* h  = uT + WN;                // [E,T,DH]

    dim3 blk(256);

    // Pre-convert x (one pass, hardware cvt)
    cvt_x_kernel<<<dim3((unsigned)(XN / (256 * 8))), blk, 0, stream>>>(x, xb);
    // Transpose+convert weights: gate/down [DIN][DH] -> [DH][DIN]
    transp_cvt_kernel<<<dim3(DH_ / 32, DIN_ / 32, E_), blk, 0, stream>>>(gate, gT, DIN_, DH_);
    transp_cvt_kernel<<<dim3(DH_ / 32, DIN_ / 32, E_), blk, 0, stream>>>(down, dT, DIN_, DH_);
    // up [DH][DIN] -> [DIN][DH]
    transp_cvt_kernel<<<dim3(DIN_ / 32, DH_ / 32, E_), blk, 0, stream>>>(up, uT, DH_, DIN_);

    // GEMMs
    moe_gatedown_kernel<<<dim3(DH_ / BN, T_ / BM, E_), blk, 0, stream>>>(xb, gT, dT, h);
    moe_up_kernel<<<dim3(DIN_ / BN, T_ / BM, E_), blk, 0, stream>>>(h, uT, out);
}
